// SpMiddleFHD_18270790877716
// MI455X (gfx1250) — matrix operand order repaired and checked by reading, not compiled, not run on hardware
//
#include <hip/hip_runtime.h>
#include <stdint.h>
#include <stddef.h>

// ---------------------------------------------------------------------------
// Types for CDNA5 WMMA (wave32): v_wmma_f32_16x16x32_bf16
// ---------------------------------------------------------------------------
typedef __bf16 v16bf __attribute__((ext_vector_type(16)));
typedef __bf16 v8bf  __attribute__((ext_vector_type(8)));
typedef float  v8f   __attribute__((ext_vector_type(8)));

static __device__ __forceinline__ unsigned short f2bf(float f) {
  unsigned int u = __float_as_uint(f);
  u += 0x7FFFu + ((u >> 16) & 1u);           // round-to-nearest-even
  return (unsigned short)(u >> 16);
}
static __device__ __forceinline__ float bf2f(unsigned short h) {
  return __uint_as_float(((unsigned int)h) << 16);
}

// ---------------------------------------------------------------------------
// Utility kernels
// ---------------------------------------------------------------------------
__global__ void fill_zero_u32(uint32_t* __restrict__ p, long n) {
  long i = (long)blockIdx.x * blockDim.x + threadIdx.x;
  long s = (long)gridDim.x * blockDim.x;
  for (; i < n; i += s) p[i] = 0u;
}

// Scatter sparse voxel features (f32 -> bf16) + build mask at full res.
__global__ void scatter_kernel(const float* __restrict__ vf,
                               const int* __restrict__ coors,
                               unsigned short* __restrict__ feat,
                               float* __restrict__ mask,
                               int N, int H, int W, int Cp) {
  int i = blockIdx.x * blockDim.x + threadIdx.x;
  if (i >= N) return;
  int z = coors[4 * i + 1], y = coors[4 * i + 2], x = coors[4 * i + 3];
  long v = ((long)z * H + y) * W + x;
  mask[v] = 1.0f;
  unsigned short* f = feat + v * Cp;
  for (int c = 0; c < 4; ++c) f[c] = f2bf(vf[4 * i + c]);
}

// Fold BN into (a,b): y = a*x + b; zero for padded channels.
__global__ void bn_pack_kernel(const float* __restrict__ g, const float* __restrict__ be,
                               const float* __restrict__ m, const float* __restrict__ v,
                               float* __restrict__ a, float* __restrict__ b,
                               int cout, int Coutp) {
  int i = blockIdx.x * blockDim.x + threadIdx.x;
  if (i >= Coutp) return;
  if (i < cout) {
    float s = g[i] * rsqrtf(v[i] + 1e-3f);
    a[i] = s;
    b[i] = be[i] - m[i] * s;
  } else {
    a[i] = 0.f;
    b[i] = 0.f;
  }
}

// Repack conv weights (kd,kh,kw,cin,cout) f32 into WMMA-B per-lane bf16 layout:
//   wp[((t*nChunks + c)*32 + L)*16 + j]  holds  column n=t*16+(L&15),
//   K = c*32 + (L>>4)*8 + (j&7) + (j>>3)*16, with K = tap*Cinp + ci. Zero-padded.
__global__ void pack_w_kernel(const float* __restrict__ w, unsigned short* __restrict__ wp,
                              int cin, int cout, int kvol, int Cinp, int Coutp, int Ktot) {
  int total = Coutp * Ktot;
  int i = blockIdx.x * blockDim.x + threadIdx.x;
  if (i >= total) return;
  int j = i & 15;
  int L = (i >> 4) & 31;
  int rest = i >> 9;                 // t*nChunks + c
  int nChunks = Ktot >> 5;
  int c = rest % nChunks;
  int t = rest / nChunks;
  int K = (c << 5) + ((L >> 4) << 3) + (j & 7) + ((j >> 3) << 4);
  int n = (t << 4) + (L & 15);
  int tap = K / Cinp;
  int ci  = K - tap * Cinp;
  float val = 0.f;
  if (tap < kvol && ci < cin && n < cout)
    val = w[((long)tap * cin + ci) * cout + n];
  wp[i] = f2bf(val);
}

// mask2[z,y,x] = any(mask[z, 2y..2y+2, 2x..2x+2]) (kernel (1,3,3), stride (1,2,2), VALID)
__global__ void mask_down_kernel(const float* __restrict__ mi, float* __restrict__ mo,
                                 int inH, int inW, int outD, int outH, int outW) {
  long total = (long)outD * outH * outW;
  long i = (long)blockIdx.x * blockDim.x + threadIdx.x;
  if (i >= total) return;
  int x = (int)(i % outW);
  long t = i / outW;
  int y = (int)(t % outH);
  int z = (int)(t / outH);
  const float* base = mi + (long)z * inH * inW;
  float acc = 0.f;
  for (int dy = 0; dy < 3; ++dy)
    for (int dx = 0; dx < 3; ++dx)
      acc += base[(long)(2 * y + dy) * inW + (2 * x + dx)];
  mo[i] = (acc > 0.f) ? 1.f : 0.f;
}

// Final: permute (D,H,W,Cp) bf16 -> (C,D,W,H) f32, C=8 actual.
__global__ void final_out_kernel(const unsigned short* __restrict__ feat,
                                 float* __restrict__ out,
                                 int D, int H, int W, int C, int Cp) {
  long total = (long)C * D * W * H;
  long i = (long)blockIdx.x * blockDim.x + threadIdx.x;
  if (i >= total) return;
  int y = (int)(i % H);
  long t = i / H;
  int x = (int)(t % W);
  t /= W;
  int z = (int)(t % D);
  int c = (int)(t / D);
  out[i] = bf2f(feat[(((long)z * H + y) * W + x) * Cp + c]);
}

// ---------------------------------------------------------------------------
// Masked implicit-GEMM 3D conv using v_wmma_f32_16x16x32_bf16.
// Fully specialized per layer shape: all loop trips / strides are constexpr.
// Block: 32 output voxels (2 M-tiles) x all N-tiles; 1 wave per (mt,nt).
// ---------------------------------------------------------------------------
template <int CINP, int COUTP, int KD, int KH, int KW, int STR, int PADZ,
          int PADHW, int TPS>
__global__ __launch_bounds__(256) void conv_wmma_t(
    const unsigned short* __restrict__ in, unsigned short* __restrict__ out,
    const float* __restrict__ maskOut,
    const unsigned short* __restrict__ wp,
    const float* __restrict__ bnA, const float* __restrict__ bnB,
    int inD, int inH, int inW, int outD, int outH, int outW) {
  constexpr int S = CINP * TPS;              // staged K halves (32 or 64)
  constexpr int ROWS = S + 8;                // LDS row stride (bank-conflict pad)
  constexpr int KVOL = KD * KH * KW;
  constexpr int KHW = KH * KW;
  constexpr int KVOLPAD = (TPS == 2) ? ((KVOL + 1) & ~1) : KVOL;
  constexpr int NSTAGES = KVOLPAD / TPS;
  constexpr int CPS = S >> 5;                // WMMA chunks per stage
  constexpr int NCHUNKS = (KVOLPAD * CINP) >> 5;
  constexpr int CSEGRT = CINP >> 3;          // 16B segments per row per tap
  constexpr int SEGTOTAL = 32 * TPS * CSEGRT;
  constexpr int NNT = COUTP >> 4;
  constexpr int BLOCK = 64 * NNT;            // 2 M-tiles x NNT waves

  __shared__ unsigned short sA[32 * ROWS];
  __shared__ float sMask[32];
  __shared__ int sCoord[32 * 4];             // z,y,x,valid per row

  const int tid = threadIdx.x;
  const int lane = tid & 31;
  const int wid = tid >> 5;
  const int mt = wid / NNT;                  // 0..1
  const int nt = wid - mt * NNT;
  const long nOut = (long)outD * outH * outW;
  const long rowBase = (long)blockIdx.x * 32;

  if (tid < 32) {
    long r = rowBase + tid;
    int z = 0, y = 0, x = 0, valid = 0;
    float mk = 0.f;
    if (r < nOut) {
      valid = 1;
      int t = (int)r;
      x = t % outW; t /= outW;
      y = t % outH; z = t / outH;
      mk = maskOut[r];
    }
    sCoord[tid * 4 + 0] = z; sCoord[tid * 4 + 1] = y;
    sCoord[tid * 4 + 2] = x; sCoord[tid * 4 + 3] = valid;
    sMask[tid] = mk;
  }
  __syncthreads();

  // Sparse early exit: all 32 output masks zero -> outputs are exactly zero.
  // Lane L tests row L; wave32 ballot is uniform across waves.
  unsigned long long bl = __ballot(sMask[lane] != 0.f);
  if (bl == 0ull) {
    long nrows = nOut - rowBase;
    if (nrows > 32) nrows = 32;
    long words = (nrows * COUTP) >> 1;       // contiguous span, b32 stores
    uint32_t* o = (uint32_t*)(out + rowBase * COUTP);
    for (long i = tid; i < words; i += BLOCK) o[i] = 0u;
    return;
  }

  v8f acc = {};
  const unsigned short* wpw = wp + (long)nt * NCHUNKS * 512;  // 32 lanes*16 halves

  for (int st = 0; st < NSTAGES; ++st) {
    __syncthreads();
    // Cooperative gather of TPS taps x 32 rows x CINP bf16 into LDS.
#pragma unroll
    for (int it = 0; it < (SEGTOTAL + BLOCK - 1) / BLOCK; ++it) {
      int s = tid + it * BLOCK;
      if (SEGTOTAL % BLOCK != 0 && s >= SEGTOTAL) break;
      int r = s / (TPS * CSEGRT);
      int rem = s - r * (TPS * CSEGRT);
      int tt = rem / CSEGRT;
      int cs = rem - tt * CSEGRT;
      int tap = st * TPS + tt;
      uint4 val = make_uint4(0u, 0u, 0u, 0u);
      if (tap < KVOL && sCoord[r * 4 + 3]) {
        int dz = tap / KHW;
        int rr = tap - dz * KHW;
        int dy = rr / KW;
        int dx = rr - dy * KW;
        int iz = sCoord[r * 4 + 0] + dz - PADZ;
        int iy = sCoord[r * 4 + 1] * STR + dy - PADHW;
        int ix = sCoord[r * 4 + 2] * STR + dx - PADHW;
        if ((unsigned)iz < (unsigned)inD && (unsigned)iy < (unsigned)inH &&
            (unsigned)ix < (unsigned)inW) {
          long off = (((long)iz * inH + iy) * (long)inW + ix) * CINP + cs * 8;
          val = *(const uint4*)(in + off);
        }
      }
      *(uint4*)(sA + r * ROWS + tt * CINP + cs * 8) = val;
    }
    __syncthreads();

    // WMMA over the 32-wide K chunks of this stage.
    const unsigned short* arow =
        sA + (mt * 16 + (lane & 15)) * ROWS + ((lane >> 4) << 3);
#pragma unroll
    for (int j = 0; j < CPS; ++j) {
      constexpr int KC0 = 0;  // silence unused in CPS==1 builds
      (void)KC0;
      int kc = j << 5;
      union { v16bf v; v8bf h[2]; } a;
      a.h[0] = *(const v8bf*)(arow + kc);        // K kc+seg .. +7
      a.h[1] = *(const v8bf*)(arow + kc + 16);   // K kc+16+seg .. +7
      int chunk = st * CPS + j;
      v16bf b = *(const v16bf*)(wpw + ((long)chunk * 32 + lane) * 16);
      acc = __builtin_amdgcn_wmma_f32_16x16x32_bf16(
          false, a.v, false, b, (short)0, acc, false, false);
    }
  }

  // Epilogue: BN + ReLU + mask, store bf16.
  const int ncol = (nt << 4) + (lane & 15);
  const float a_ = bnA[ncol];
  const float b_ = bnB[ncol];
  const int mbase = (mt << 4) + ((lane >> 4) << 3);
#pragma unroll
  for (int r = 0; r < 8; ++r) {
    int mrow = mbase + r;
    long row = rowBase + mrow;
    if (row < nOut) {
      float vv = fmaxf(acc[r] * a_ + b_, 0.f) * sMask[mrow];
      out[row * COUTP + ncol] = f2bf(vv);
    }
  }
}

// ---------------------------------------------------------------------------
// Host orchestration
// ---------------------------------------------------------------------------
struct LayerDesc {
  int cin, cout, kd, kh, kw, strH, strW, padD, padH, padW;
};
static const LayerDesc LS[15] = {
    {4, 8, 3, 3, 3, 1, 1, 1, 1, 1},   {8, 8, 3, 3, 3, 1, 1, 1, 1, 1},
    {8, 16, 3, 3, 3, 1, 1, 1, 1, 1},  {16, 16, 3, 3, 3, 1, 1, 1, 1, 1},
    {16, 32, 3, 3, 3, 1, 1, 1, 1, 1}, {32, 32, 3, 3, 3, 1, 1, 1, 1, 1},
    {32, 64, 3, 3, 3, 1, 1, 1, 1, 1}, {64, 64, 3, 3, 3, 1, 1, 1, 1, 1},
    {64, 64, 1, 3, 3, 2, 2, 0, 0, 0}, {64, 32, 3, 3, 3, 1, 1, 1, 1, 1},
    {32, 32, 3, 3, 3, 1, 1, 1, 1, 1}, {32, 16, 3, 3, 3, 1, 1, 1, 1, 1},
    {16, 16, 3, 3, 3, 1, 1, 1, 1, 1}, {16, 8, 3, 3, 3, 1, 1, 1, 1, 1},
    {8, 8, 3, 3, 3, 1, 1, 1, 1, 1}};

extern "C" void kernel_launch(void* const* d_in, const int* in_sizes, int n_in,
                              void* d_out, int out_size, void* d_ws, size_t ws_size,
                              hipStream_t stream) {
  (void)out_size;
  const int FD = 24, FH = 201, FW = 177;          // full-res grid
  const int DD = 24, DH = 100, DW = 88;           // after layer-8 downsample
  const long FULLVOX = (long)FD * FH * FW;        // 853,848
  const long DOWNVOX = (long)DD * DH * DW;        // 211,200

  // Locate flattened params (input_shape tuple may flatten as 1 or 3 leaves).
  int wbase = -1;
  for (int i = 2; i < n_in; ++i)
    if (in_sizes[i] == 3 * 3 * 3 * 4 * 8) { wbase = i; break; }
  if (wbase < 0 || wbase + 15 + 60 > n_in) wbase = 4;
  const int bnbase = wbase + 15;

  const float* vf = (const float*)d_in[0];
  const int* coors = (const int*)d_in[1];
  const int N = in_sizes[0] / 4;

  // Workspace layout.
  auto al = [](size_t v) { return (v + 255) & ~(size_t)255; };
  size_t off = 0;
  size_t F0 = off; off = al(off + (size_t)FULLVOX * 64 * 2);
  size_t F1 = off; off = al(off + (size_t)FULLVOX * 64 * 2);
  size_t MF = off; off = al(off + (size_t)FULLVOX * 4);
  size_t MD = off; off = al(off + (size_t)DOWNVOX * 4);
  size_t BN = off; off = al(off + 15 * 512);      // per layer: 64 a + 64 b f32
  size_t WP[15];
  int Cinp_[15], Coutp_[15], TPS_[15], kvolPad_[15], Ktot_[15];
  for (int i = 0; i < 15; ++i) {
    const LayerDesc& L = LS[i];
    int Cinp = L.cin < 16 ? 16 : L.cin;
    int Coutp = L.cout < 16 ? 16 : L.cout;
    int kvol = L.kd * L.kh * L.kw;
    int TPS = (Cinp == 16) ? 2 : 1;
    int kvolPad = (TPS == 2) ? ((kvol + 1) & ~1) : kvol;
    Cinp_[i] = Cinp; Coutp_[i] = Coutp; TPS_[i] = TPS;
    kvolPad_[i] = kvolPad; Ktot_[i] = kvolPad * Cinp;
    WP[i] = off;
    off = al(off + (size_t)Coutp * Ktot_[i] * 2);
  }
  if (off > ws_size) return;                      // workspace too small
  char* ws = (char*)d_ws;

  // 1) Zero initial feature buffer (Cp=16) and full-res mask.
  fill_zero_u32<<<4096, 256, 0, stream>>>((uint32_t*)(ws + F0), FULLVOX * 8);
  fill_zero_u32<<<4096, 256, 0, stream>>>((uint32_t*)(ws + MF), FULLVOX);

  // 2) Scatter voxel features.
  scatter_kernel<<<(N + 255) / 256, 256, 0, stream>>>(
      vf, coors, (unsigned short*)(ws + F0), (float*)(ws + MF), N, FH, FW, 16);

  // 3) Fold BN and repack weights for every layer.
  for (int i = 0; i < 15; ++i) {
    const LayerDesc& L = LS[i];
    float* bnA = (float*)(ws + BN + (size_t)i * 512);
    float* bnB = bnA + 64;
    bn_pack_kernel<<<1, 64, 0, stream>>>(
        (const float*)d_in[bnbase + 4 * i + 0], (const float*)d_in[bnbase + 4 * i + 1],
        (const float*)d_in[bnbase + 4 * i + 2], (const float*)d_in[bnbase + 4 * i + 3],
        bnA, bnB, L.cout, Coutp_[i]);
    int total = Coutp_[i] * Ktot_[i];
    pack_w_kernel<<<(total + 255) / 256, 256, 0, stream>>>(
        (const float*)d_in[wbase + i], (unsigned short*)(ws + WP[i]),
        L.cin, L.cout, L.kd * L.kh * L.kw, Cinp_[i], Coutp_[i], Ktot_[i]);
  }

  // 4) Conv chain (ping-pong F0 <-> F1).
  size_t cur = F0, nxt = F1;
  for (int i = 0; i < 15; ++i) {
    int inD, inH, inW, outD, outH, outW;
    if (i < 8) { inD = FD; inH = FH; inW = FW; outD = FD; outH = FH; outW = FW; }
    else if (i == 8) { inD = FD; inH = FH; inW = FW; outD = DD; outH = DH; outW = DW; }
    else { inD = DD; inH = DH; inW = DW; outD = DD; outH = DH; outW = DW; }

    if (i == 8) {
      mask_down_kernel<<<(int)((DOWNVOX + 255) / 256), 256, 0, stream>>>(
          (const float*)(ws + MF), (float*)(ws + MD), FH, FW, DD, DH, DW);
    }
    const float* maskOut = (i < 8) ? (const float*)(ws + MF) : (const float*)(ws + MD);
    const float* bnA = (const float*)(ws + BN + (size_t)i * 512);
    const float* bnB = bnA + 64;
    const unsigned short* cin_p = (const unsigned short*)(ws + cur);
    unsigned short* cout_p = (unsigned short*)(ws + nxt);
    const unsigned short* wpp = (const unsigned short*)(ws + WP[i]);

    long nOut = (long)outD * outH * outW;
    int grid = (int)((nOut + 31) / 32);

#define CONV_ARGS cin_p, cout_p, maskOut, wpp, bnA, bnB, inD, inH, inW, outD, outH, outW
    switch (i) {
      case 0: case 1: case 2: case 3: case 12: case 13: case 14:
        conv_wmma_t<16, 16, 3, 3, 3, 1, 1, 1, 2><<<grid, 64, 0, stream>>>(CONV_ARGS);
        break;
      case 4:
        conv_wmma_t<16, 32, 3, 3, 3, 1, 1, 1, 2><<<grid, 128, 0, stream>>>(CONV_ARGS);
        break;
      case 5: case 10:
        conv_wmma_t<32, 32, 3, 3, 3, 1, 1, 1, 1><<<grid, 128, 0, stream>>>(CONV_ARGS);
        break;
      case 6:
        conv_wmma_t<32, 64, 3, 3, 3, 1, 1, 1, 1><<<grid, 256, 0, stream>>>(CONV_ARGS);
        break;
      case 7:
        conv_wmma_t<64, 64, 3, 3, 3, 1, 1, 1, 1><<<grid, 256, 0, stream>>>(CONV_ARGS);
        break;
      case 8:
        conv_wmma_t<64, 64, 1, 3, 3, 2, 0, 0, 1><<<grid, 256, 0, stream>>>(CONV_ARGS);
        break;
      case 9:
        conv_wmma_t<64, 32, 3, 3, 3, 1, 1, 1, 1><<<grid, 128, 0, stream>>>(CONV_ARGS);
        break;
      case 11:
        conv_wmma_t<32, 16, 3, 3, 3, 1, 1, 1, 1><<<grid, 64, 0, stream>>>(CONV_ARGS);
        break;
    }
#undef CONV_ARGS
    size_t t = cur; cur = nxt; nxt = t;
  }

  // 5) Final permuted f32 output: (C=8, D, W, H).
  long total = 8L * DD * DW * DH;
  final_out_kernel<<<(int)((total + 255) / 256), 256, 0, stream>>>(
      (const unsigned short*)(ws + cur), (float*)d_out, DD, DH, DW, 8, Coutp_[14]);
}
